// SelfAttention_53377853555278
// MI455X (gfx1250) — matrix operand order repaired and checked by reading, not compiled, not run on hardware
//
#include <hip/hip_runtime.h>
#include <hip/hip_bf16.h>

// ---------------------------------------------------------------------------
// CDNA5 (gfx1250) wave32 WMMA self-attention (SAGAN block)
//   B=4, C=256, Ck=32, N=64*64=4096
// Pipeline:
//   k0: convert Wf/Wg/Wh f32 -> f16
//   k1: transpose-convert x [B,C,N] f32 -> xT [B,N,C] f16 (LDS tiled)
//   k2: projections via v_wmma_f32_16x16x32_f16
//   k3: softmax row stats t_j = m_j + ln(sum_i exp(S[j,i]-m_j))
//   k4: out[c,i] = gamma * sum_j h[c,j]*exp(S[j,i]-t_j) + x[c,i]
//       h tiles staged into LDS by the Tensor Data Mover (double buffered,
//       s_wait_tensorcnt paced), WMMA A-operands read back via ds_load.
// ---------------------------------------------------------------------------

typedef __attribute__((ext_vector_type(16))) _Float16 v16h;
typedef __attribute__((ext_vector_type(8)))  float    v8f;
typedef __attribute__((ext_vector_type(4)))  unsigned u32x4;
typedef __attribute__((ext_vector_type(2)))  unsigned u32x2;
typedef __attribute__((ext_vector_type(4)))  unsigned tdm_g0_t;
typedef __attribute__((ext_vector_type(8)))  int      tdm_g1_t;
typedef __attribute__((ext_vector_type(4)))  int      tdm_g2_t;

#define NB   4096   // N pixels
#define CC   256    // channels
#define CK   32     // key/query channels

// LDS row pitch for staged h tiles: 32 halves + 2-DWORD TDM pad = 72 bytes.
#define LDH        36
#define HBUF_BYTES (256 * LDH * 2)          // 18432
#define SMEM_BYTES (2 * HBUF_BYTES + NB*4)  // 2 bufs + 16KB stats = 53248

#if defined(__has_builtin)
#  if __has_builtin(__builtin_amdgcn_tensor_load_to_lds) && \
      __has_builtin(__builtin_amdgcn_s_wait_tensorcnt)
#    define HAVE_TDM 1
#  endif
#endif
#ifndef HAVE_TDM
#  define HAVE_TDM 0
#endif

__device__ __forceinline__ v8f wmma16(v16h a, v16h b, v8f c) {
  return __builtin_amdgcn_wmma_f32_16x16x32_f16(
      false, a, false, b, (short)0, c, false, false);
}

// A operand: 16(M) x 32(K) f16 from row-major storage, row stride ld (halves).
__device__ __forceinline__ v16h loadA(const _Float16* tile, int ld, int lane) {
  const _Float16* p = tile + (size_t)(lane & 15) * ld + ((lane & 16) ? 16 : 0);
  u32x4 lo = *(const u32x4*)(p);
  u32x4 hi = *(const u32x4*)(p + 8);
  v16h a;
  unsigned* w = (unsigned*)&a;
  w[0] = lo.x; w[1] = lo.y; w[2] = lo.z; w[3] = lo.w;
  w[4] = hi.x; w[5] = hi.y; w[6] = hi.z; w[7] = hi.w;
  return a;
}

// A operand from LDS tile with LDH-half row pitch (8B-aligned b64 chunks).
__device__ __forceinline__ v16h loadA_lds(const _Float16* tile, int lane) {
  const _Float16* p = tile + (lane & 15) * LDH + ((lane & 16) ? 16 : 0);
  u32x2 q0 = *(const u32x2*)(p);
  u32x2 q1 = *(const u32x2*)(p + 4);
  u32x2 q2 = *(const u32x2*)(p + 8);
  u32x2 q3 = *(const u32x2*)(p + 12);
  v16h a;
  unsigned* w = (unsigned*)&a;
  w[0] = q0.x; w[1] = q0.y; w[2] = q1.x; w[3] = q1.y;
  w[4] = q2.x; w[5] = q2.y; w[6] = q3.x; w[7] = q3.y;
  return a;
}

// B operand: 32(K) x 16(N) f16 from [n][k] storage (K contiguous per column).
__device__ __forceinline__ v16h loadB(const _Float16* colBase, int ldcol, int lane) {
  const _Float16* p = colBase + (size_t)(lane & 15) * ldcol + ((lane & 16) ? 16 : 0);
  u32x4 lo = *(const u32x4*)(p);
  u32x4 hi = *(const u32x4*)(p + 8);
  v16h b;
  unsigned* w = (unsigned*)&b;
  w[0] = lo.x; w[1] = lo.y; w[2] = lo.z; w[3] = lo.w;
  w[4] = hi.x; w[5] = hi.y; w[6] = hi.z; w[7] = hi.w;
  return b;
}

__device__ __forceinline__ unsigned pack2h(float a, float b) {
  union { _Float16 h[2]; unsigned u; } z;
  z.h[0] = (_Float16)a; z.h[1] = (_Float16)b;
  return z.u;
}

#if HAVE_TDM
// Issue TDM load of h[0:256][j0:j0+32] (f16, row stride 4096) into LDS at
// lds_addr, padded to LDH-half rows (pad_interval=16 DW, pad_amount=2 DW).
__device__ __forceinline__ void tdm_stage_h(const _Float16* gsrc, unsigned lds_addr) {
  unsigned long long ga = (unsigned long long)gsrc;
  unsigned ga_lo = (unsigned)__builtin_amdgcn_readfirstlane((int)(ga & 0xffffffffull));
  unsigned ga_hi = (unsigned)__builtin_amdgcn_readfirstlane((int)(ga >> 32));
  unsigned lds_u = (unsigned)__builtin_amdgcn_readfirstlane((int)lds_addr);
  tdm_g0_t g0 = { 1u,                                    // count=1 (valid D#)
                  lds_u,                                 // lds_addr (bytes)
                  ga_lo,                                 // global_addr[31:0]
                  (ga_hi & 0x01ffffffu) | (2u << 30) };  // ga[56:32] | type=2
  tdm_g1_t g1 = { (int)((1u << 16)        // data_size = 2 bytes
                      | (1u << 20)        // pad_enable
                      | (3u << 22)        // pad_interval = 16 DWORDs
                      | (1u << 25)),      // pad_amount  = 2 DWORDs
                  (int)(4096u << 16),     // tensor_dim0[15:0] in [31:16]
                  (int)(256u  << 16),     // tensor_dim0[31:16]=0 | tensor_dim1[15:0]
                  (int)(32u   << 16),     // tensor_dim1[31:16]=0 | tile_dim0=32
                  (int)(256u),            // tile_dim1=256 | tile_dim2=0
                  (int)(4096u),           // tensor_dim0_stride[31:0]
                  0, 0 };                 // stride hi / dim1_stride = 0
  tdm_g2_t gz = { 0, 0, 0, 0 };
#if __clang_major__ >= 23
  tdm_g1_t gz8 = { 0, 0, 0, 0, 0, 0, 0, 0 };
  __builtin_amdgcn_tensor_load_to_lds(g0, g1, gz, gz, gz8, 0);
#else
  __builtin_amdgcn_tensor_load_to_lds(g0, g1, gz, gz, 0);
#endif
}
#else
// Fallback: cooperative copy, 128 threads, 2 rows each, LDH-half pitch.
__device__ __forceinline__ void stage_h_fallback(const _Float16* gsrc,
                                                 _Float16* buf, int tid) {
  for (int r = tid; r < 256; r += 128) {
    const u32x4* s = (const u32x4*)(gsrc + (size_t)r * NB);
    _Float16* d = buf + r * LDH;
#pragma unroll
    for (int q = 0; q < 4; ++q) {
      u32x4 v = s[q];
      u32x2 a = { v.x, v.y };
      u32x2 b = { v.z, v.w };
      *(u32x2*)(d + q * 8)     = a;
      *(u32x2*)(d + q * 8 + 4) = b;
    }
  }
}
#endif

// ---------------- k0: weights f32 -> f16 -----------------------------------
__global__ void sa_convert_weights(const float* __restrict__ Wf,
                                   const float* __restrict__ Wg,
                                   const float* __restrict__ Wh,
                                   _Float16* __restrict__ Wf16,
                                   _Float16* __restrict__ Wg16,
                                   _Float16* __restrict__ Wh16) {
  int i = blockIdx.x * 256 + threadIdx.x;
  if (i < CK * CC) { Wf16[i] = (_Float16)Wf[i]; Wg16[i] = (_Float16)Wg[i]; }
  if (i < CC * CC) { Wh16[i] = (_Float16)Wh[i]; }
}

// ---------------- k1: x [B,C,N] f32 -> xT [B,N,C] f16 (tiled) ---------------
__global__ void sa_transpose_x(const float* __restrict__ x,
                               _Float16* __restrict__ xT) {
  __shared__ float tile[32][33];
  int blk  = blockIdx.x;            // 4*128*8 = 4096 blocks
  int cblk = blk & 7;
  int nblk = (blk >> 3) & 127;
  int b    = blk >> 10;
  int tx = threadIdx.x & 31;
  int ty = threadIdx.x >> 5;
  const float* xp = x + ((size_t)(b * CC + cblk * 32)) * NB + nblk * 32;
#pragma unroll
  for (int r = 0; r < 32; r += 8)
    tile[ty + r][tx] = xp[(size_t)(ty + r) * NB + tx];
  __syncthreads();
  _Float16* xo = xT + ((size_t)(b * NB + nblk * 32)) * CC + cblk * 32;
#pragma unroll
  for (int r = 0; r < 32; r += 8)
    xo[(size_t)(ty + r) * CC + tx] = (_Float16)tile[tx][ty + r];
}

// ---------------- k2: projections f,g,h via WMMA ----------------------------
__global__ void sa_proj(const _Float16* __restrict__ xT,
                        const _Float16* __restrict__ Wh16,
                        const _Float16* __restrict__ Wf16,
                        const _Float16* __restrict__ Wg16,
                        const float* __restrict__ bh,
                        const float* __restrict__ bf,
                        const float* __restrict__ bg,
                        _Float16* __restrict__ h16,
                        _Float16* __restrict__ fT,
                        _Float16* __restrict__ gT) {
  int blk  = blockIdx.x;            // B*(N/16) = 1024
  int b    = blk >> 8;
  int n0   = (blk & 255) << 4;
  int wave = threadIdx.x >> 5;      // wave-uniform
  int lane = threadIdx.x & 31;
  int n    = lane & 15;
  int hi   = lane >> 4;

  v8f zero = {};
  v8f acc[4];  v8f afg[2];
#pragma unroll
  for (int t = 0; t < 4; ++t) acc[t] = zero;
  afg[0] = zero; afg[1] = zero;

  const _Float16* xTb = xT + (size_t)(b * NB + n0) * CC;

  for (int c0 = 0; c0 < CC; c0 += 32) {
    v16h bx = loadB(xTb + c0, CC, lane);
#pragma unroll
    for (int t = 0; t < 4; ++t) {
      v16h aw = loadA(Wh16 + (size_t)(wave * 64 + t * 16) * CC + c0, CC, lane);
      acc[t] = wmma16(aw, bx, acc[t]);
    }
    if (wave == 0) {
#pragma unroll
      for (int t = 0; t < 2; ++t) {
        v16h aw = loadA(Wf16 + (size_t)(t * 16) * CC + c0, CC, lane);
        afg[t] = wmma16(aw, bx, afg[t]);
      }
    } else if (wave == 1) {
#pragma unroll
      for (int t = 0; t < 2; ++t) {
        v16h aw = loadA(Wg16 + (size_t)(t * 16) * CC + c0, CC, lane);
        afg[t] = wmma16(aw, bx, afg[t]);
      }
    }
  }

#pragma unroll
  for (int t = 0; t < 4; ++t) {
#pragma unroll
    for (int v = 0; v < 8; ++v) {
      int c = wave * 64 + t * 16 + hi * 8 + v;
      float val = acc[t][v] + bh[c];
      h16[(size_t)(b * CC + c) * NB + n0 + n] = (_Float16)val;
    }
  }
  if (wave < 2) {
    const float* bias  = (wave == 0) ? bf : bg;
    _Float16*    dst   = (wave == 0) ? fT : gT;
#pragma unroll
    for (int t = 0; t < 2; ++t) {
      unsigned pk[4];
#pragma unroll
      for (int v = 0; v < 8; v += 2) {
        int k = t * 16 + hi * 8 + v;
        pk[v >> 1] = pack2h(afg[t][v] + bias[k], afg[t][v + 1] + bias[k + 1]);
      }
      u32x4 st = {pk[0], pk[1], pk[2], pk[3]};
      *(u32x4*)(dst + (size_t)(b * NB + n0 + n) * CK + t * 16 + hi * 8) = st;
    }
  }
}

// ---------------- k3: softmax row stats t_j = m_j + ln(l_j) -----------------
__global__ void sa_stats(const _Float16* __restrict__ fT,
                         const _Float16* __restrict__ gT,
                         float* __restrict__ tstat) {
  int blk  = blockIdx.x;            // B*(N/64) = 256
  int b    = blk >> 6;
  int wave = threadIdx.x >> 5;
  int j0   = ((blk & 63) * 4 + wave) * 16;
  int lane = threadIdx.x & 31;
  int hi   = lane >> 4;

  v16h aq = loadA(fT + (size_t)(b * NB + j0) * CK, CK, lane);

  float m[8], s[8];
#pragma unroll
  for (int v = 0; v < 8; ++v) { m[v] = -1.0e30f; s[v] = 0.0f; }

  for (int i0 = 0; i0 < NB; i0 += 16) {
    v16h bk = loadB(gT + (size_t)(b * NB + i0) * CK, CK, lane);
    v8f c = {};
    c = wmma16(aq, bk, c);
#pragma unroll
    for (int v = 0; v < 8; ++v) {
      float sv = c[v];
      float mn = fmaxf(m[v], sv);
      s[v] = s[v] * __expf(m[v] - mn) + __expf(sv - mn);
      m[v] = mn;
    }
  }
#pragma unroll
  for (int off = 1; off < 16; off <<= 1) {
#pragma unroll
    for (int v = 0; v < 8; ++v) {
      float mo = __shfl_xor(m[v], off, 32);
      float so = __shfl_xor(s[v], off, 32);
      float mn = fmaxf(m[v], mo);
      s[v] = s[v] * __expf(m[v] - mn) + so * __expf(mo - mn);
      m[v] = mn;
    }
  }
  if ((lane & 15) == 0) {
#pragma unroll
    for (int v = 0; v < 8; ++v)
      tstat[(size_t)b * NB + j0 + hi * 8 + v] = m[v] + __logf(s[v]);
  }
}

// ---------------- k4: out = gamma * h @ P + x -------------------------------
// grid: B*(N/64)=256 blocks, 128 threads; wave -> 16 output pixels, 256 chans.
// h tiles staged into LDS via TDM (double buffered) or cooperative fallback.
__global__ void sa_output(const _Float16* __restrict__ fT,
                          const _Float16* __restrict__ gT,
                          const _Float16* __restrict__ h16,
                          const float* __restrict__ tstat,
                          const float* __restrict__ x,
                          const float* __restrict__ gamma,
                          float* __restrict__ out) {
  __shared__ __align__(16) char smem[SMEM_BYTES];
  _Float16* hbuf0 = (_Float16*)(smem);
  _Float16* hbuf1 = (_Float16*)(smem + HBUF_BYTES);
  float*    tl    = (float*)(smem + 2 * HBUF_BYTES);

  int blk  = blockIdx.x;
  int b    = blk >> 6;
  int wave = threadIdx.x >> 5;      // wave-uniform
  int i0   = ((blk & 63) * 4 + wave) * 16;
  int lane = threadIdx.x & 31;
  int n    = lane & 15;
  int hi   = lane >> 4;

  for (int idx = threadIdx.x; idx < NB; idx += 128)
    tl[idx] = tstat[(size_t)b * NB + idx];

  v16h bq = loadB(gT + (size_t)(b * NB + i0) * CK, CK, lane);

  v8f zero = {};
  v8f acc[16];
#pragma unroll
  for (int t = 0; t < 16; ++t) acc[t] = zero;

  const _Float16* fTb = fT + (size_t)b * NB * CK;
  const _Float16* hbg = h16 + (size_t)b * CC * NB;

#if HAVE_TDM
  // low 32 bits of the generic shared-pointer == LDS byte offset
  unsigned lds_base = (unsigned)(unsigned long long)(void*)smem;
  if (wave == 0) {
    tdm_stage_h(hbg, lds_base);                      // tile j0=0 -> hbuf0
    __builtin_amdgcn_s_wait_tensorcnt(0);
  }
  __syncthreads();

  int cur = 0;
  for (int j0 = 0; j0 < NB; j0 += 32) {
    if (wave == 0 && j0 + 32 < NB)                   // prefetch next tile
      tdm_stage_h(hbg + j0 + 32, lds_base + (cur ? 0u : (unsigned)HBUF_BYTES));
    const _Float16* hb = cur ? hbuf1 : hbuf0;
#else
  for (int j0 = 0; j0 < NB; j0 += 32) {
    __syncthreads();                                 // readers of prev tile done
    stage_h_fallback(hbg + j0, hbuf0, threadIdx.x);
    __syncthreads();
    const _Float16* hb = hbuf0;
#endif
    __builtin_prefetch(fTb + (size_t)(j0 + 32) * CK, 0, 1);
    v16h a0 = loadA(fTb + (size_t)(j0)      * CK, CK, lane);
    v16h a1 = loadA(fTb + (size_t)(j0 + 16) * CK, CK, lane);
    v8f s0 = {}; s0 = wmma16(a0, bq, s0);            // S rows j0..j0+15
    v8f s1 = {}; s1 = wmma16(a1, bq, s1);            // S rows j0+16..j0+31

    // P = exp(S - t_j); pack own 8 rows of each tile as f16 pairs
    unsigned own0[4], own1[4];
#pragma unroll
    for (int v = 0; v < 8; v += 2) {
      int j = j0 + hi * 8 + v;
      float p0a = __expf(s0[v]     - tl[j]);
      float p0b = __expf(s0[v + 1] - tl[j + 1]);
      float p1a = __expf(s1[v]     - tl[j + 16]);
      float p1b = __expf(s1[v + 1] - tl[j + 17]);
      own0[v >> 1] = pack2h(p0a, p0b);
      own1[v >> 1] = pack2h(p1a, p1b);
    }
    // D layout -> B operand layout: exchange halves between lane and lane^16.
    v16h bp;
    unsigned* w = (unsigned*)&bp;
#pragma unroll
    for (int q = 0; q < 4; ++q) {
      unsigned snd = hi ? own0[q] : own1[q];
      unsigned rcv = (unsigned)__shfl_xor((int)snd, 16, 32);
      w[q]     = hi ? rcv : own0[q];   // K 16h+0..7  (hi=0: tile0 rows, hi=1: tile1 rows)
      w[4 + q] = hi ? own1[q] : rcv;   // K 16h+8..15 (hi=0: tile0 rows, hi=1: tile1 rows)
    }
    // accumulate: out[c, i] += h[c, j0..j0+31] * P  (A operand from LDS)
#pragma unroll
    for (int t = 0; t < 16; ++t) {
      v16h ah = loadA_lds(hb + t * 16 * LDH, lane);
      acc[t] = wmma16(ah, bp, acc[t]);
    }
#if HAVE_TDM
    if (wave == 0) __builtin_amdgcn_s_wait_tensorcnt(0);
    __syncthreads();
    cur ^= 1;
  }
#else
  }
#endif

  float g = gamma[0];
#pragma unroll
  for (int t = 0; t < 16; ++t) {
#pragma unroll
    for (int v = 0; v < 8; ++v) {
      int c = t * 16 + hi * 8 + v;
      size_t idx = (size_t)(b * CC + c) * NB + i0 + n;
      out[idx] = g * acc[t][v] + x[idx];
    }
  }
}

// ---------------------------------------------------------------------------
extern "C" void kernel_launch(void* const* d_in, const int* in_sizes, int n_in,
                              void* d_out, int out_size, void* d_ws, size_t ws_size,
                              hipStream_t stream) {
  const float* x     = (const float*)d_in[0];
  const float* Wf    = (const float*)d_in[1];
  const float* bf    = (const float*)d_in[2];
  const float* Wg    = (const float*)d_in[3];
  const float* bg    = (const float*)d_in[4];
  const float* Wh    = (const float*)d_in[5];
  const float* bh    = (const float*)d_in[6];
  const float* gamma = (const float*)d_in[7];
  float* out = (float*)d_out;

  char* ws = (char*)d_ws;
  // workspace layout (bytes), all 256B-aligned; total ~19.1 MB
  _Float16* Wh16 = (_Float16*)(ws + 0);                     // 131072
  _Float16* Wf16 = (_Float16*)(ws + 131072);                // 16384
  _Float16* Wg16 = (_Float16*)(ws + 147456);                // 16384
  _Float16* xT   = (_Float16*)(ws + 163840);                // 8388608
  _Float16* fT   = (_Float16*)(ws + 8552448);               // 1048576
  _Float16* gT   = (_Float16*)(ws + 9601024);               // 1048576
  _Float16* h16  = (_Float16*)(ws + 10649600);              // 8388608
  float*    tst  = (float*)   (ws + 19038208);              // 65536

  sa_convert_weights<<<256, 256, 0, stream>>>(Wf, Wg, Wh, Wf16, Wg16, Wh16);
  sa_transpose_x<<<4096, 256, 0, stream>>>(x, xT);
  sa_proj<<<1024, 128, 0, stream>>>(xT, Wh16, Wf16, Wg16, bh, bf, bg, h16, fT, gT);
  sa_stats<<<256, 128, 0, stream>>>(fT, gT, tst);
  sa_output<<<256, 128, 0, stream>>>(fT, gT, h16, tst, x, gamma, out);
}
